// ChildMaskedPredictor_37967510897250
// MI455X (gfx1250) — matrix operand order repaired and checked by reading, not compiled, not run on hardware
//
#include <hip/hip_runtime.h>
#include <hip/hip_bf16.h>

#define STATE_DIM 128
#define FEAT      64
#define NEG_INF   (-1.0e9f)
#define MIN_LS    (-6.9f)
#define MAX_LS    (-4.6f)
#define SCHUNK    32          // s values per wave (128 / 4 s-quarters)

typedef __attribute__((ext_vector_type(16))) _Float16 v16h;
typedef __attribute__((ext_vector_type(8)))  _Float16 v8h;
typedef __attribute__((ext_vector_type(8)))  float    v8f;
typedef __attribute__((ext_vector_type(4)))  float    v4f;

// ---------------- workspace layout (in _Float16 units) ----------------
#define OFF_W2   0            // [128][8 frag][32 lane][16]
#define OFF_WA1  524288       // [8][32][16]
#define OFF_WA2  528384
#define OFF_WH1  532480
#define OFF_WH2  536576
#define OFF_WMLS 540672       // [2][32][16]
#define OFF_W1   541696       // [128][64]
#define OFF_B1   549888       // [128][64]
#define OFF_B2   558080       // [128][64]
#define WS_HALFS 566272

static __device__ inline v8f wmma_f16(v16h a, v16h b, v8f c) {
  return __builtin_amdgcn_wmma_f32_16x16x32_f16(false, a, false, b, (short)0, c, false, false);
}

static __device__ inline v16h zero16h() {
  v16h z;
#pragma unroll
  for (int i = 0; i < 16; ++i) z[i] = (_Float16)0.0f;
  return z;
}

static __device__ inline v8h zero8h() {
  v8h z;
#pragma unroll
  for (int i = 0; i < 8; ++i) z[i] = (_Float16)0.0f;
  return z;
}

// A fragment from an f16 row: halves 0..7 = row[k0..k0+7], 8..15 = row[k0+16..k0+23].
static __device__ inline v16h load_afrag_h(const _Float16* row, int k0) {
  v8h x = *(const v8h*)(row + k0);
  v8h y = *(const v8h*)(row + k0 + 16);
  v16h a;
#pragma unroll
  for (int i = 0; i < 8; ++i) { a[i] = x[i]; a[8 + i] = y[i]; }
  return a;
}

// A fragment from an f32 row (only used for a_t, once per wave).
static __device__ inline v16h load_afrag_row_f32(const float* __restrict__ row, int k0) {
  const v4f* q0 = (const v4f*)(row + k0);
  const v4f* q1 = (const v4f*)(row + k0 + 16);
  v4f x0 = q0[0], x1 = q0[1], x2 = q1[0], x3 = q1[1];
  v16h a;
#pragma unroll
  for (int i = 0; i < 4; ++i) {
    a[i]      = (_Float16)x0[i];
    a[4 + i]  = (_Float16)x1[i];
    a[8 + i]  = (_Float16)x2[i];
    a[12 + i] = (_Float16)x3[i];
  }
  return a;
}

// cvt f32 tile to f16 and add broadcast bias (packed): cvt_pk + pk_add
static __device__ inline v8h cvt_add_h(v8f c, _Float16 bv) {
  v8h r, bs;
#pragma unroll
  for (int j = 0; j < 8; ++j) { r[j] = (_Float16)c[j]; bs[j] = bv; }
  return r + bs;
}

// One 16x16 output tile with bias-in-C (cold paths only).
static __device__ inline v8f tile_gemm(v16h a0, v16h a1, const _Float16* __restrict__ Wfrag,
                                       int t, int lane, float bv) {
  v16h b0 = *(const v16h*)(Wfrag + ((size_t)(t)     * 32 + lane) * 16);
  v16h b1 = *(const v16h*)(Wfrag + ((size_t)(4 + t) * 32 + lane) * 16);
  v8f c;
#pragma unroll
  for (int j = 0; j < 8; ++j) c[j] = bv;
  c = wmma_f16(a0, b0, c);
  c = wmma_f16(a1, b1, c);
  return c;
}

#define LDS_STRIDE 72  // halves per row; 144B pitch: conflict-free, keeps 16B alignment

static __device__ inline void layer64_to_lds(v16h a0, v16h a1, const _Float16* __restrict__ Wfrag,
                                             const float* __restrict__ bias,
                                             _Float16* lds, int nloc, int hi, int mrow, int lane) {
#pragma unroll
  for (int t = 0; t < 4; ++t) {
    const int g = t * 16 + nloc;
    v8f c = tile_gemm(a0, a1, Wfrag, t, lane, bias[g]);
#pragma unroll
    for (int j = 0; j < 8; ++j)
      lds[(mrow + j) * LDS_STRIDE + g] = (_Float16)fmaxf(c[j], 0.0f);
  }
}

// ---- hot-loop step: 8 B fragments as named locals, zero-C WMMA,
//      packed-f16 pooling with deferred relu, WGP$ prefetch 2 ahead ----
static __device__ __forceinline__ void s_step(
    int s, const _Float16* __restrict__ W2h, const _Float16* __restrict__ w1h,
    const _Float16* __restrict__ b1h, const _Float16* __restrict__ b2h,
    const float* __restrict__ srow, bool keep,
    int abase, int nloc, int lane, const v16h& zz, v8h (&pooled)[4])
{
  const _Float16* fb = W2h + (size_t)s * 4096;
  v16h B0 = *(const v16h*)(fb + ((size_t)(0 * 32 + lane)) * 16);
  v16h B1 = *(const v16h*)(fb + ((size_t)(1 * 32 + lane)) * 16);
  v16h B2 = *(const v16h*)(fb + ((size_t)(2 * 32 + lane)) * 16);
  v16h B3 = *(const v16h*)(fb + ((size_t)(3 * 32 + lane)) * 16);
  v16h B4 = *(const v16h*)(fb + ((size_t)(4 * 32 + lane)) * 16);
  v16h B5 = *(const v16h*)(fb + ((size_t)(5 * 32 + lane)) * 16);
  v16h B6 = *(const v16h*)(fb + ((size_t)(6 * 32 + lane)) * 16);
  v16h B7 = *(const v16h*)(fb + ((size_t)(7 * 32 + lane)) * 16);

  // pull s+2's 8KB fragment block toward WGP$ (stays inside ws)
  const char* pf = (const char*)(W2h + (size_t)(s + 2) * 4096);
  __builtin_prefetch(pf + lane * 256, 0, 1);
  __builtin_prefetch(pf + lane * 256 + 128, 0, 1);

  // h1 = relu(x*w1 + b1) in packed f16 (x broadcast via op_sel)
  const _Float16 xh = (_Float16)srow[s];
  v16h xs;
#pragma unroll
  for (int i = 0; i < 16; ++i) xs[i] = xh;
  const _Float16* w1r = w1h + s * FEAT;
  const _Float16* b1r = b1h + s * FEAT;
  v16h a0 = load_afrag_h(w1r, 0 + abase)  * xs + load_afrag_h(b1r, 0 + abase);
  v16h a1 = load_afrag_h(w1r, 32 + abase) * xs + load_afrag_h(b1r, 32 + abase);
  a0 = __builtin_elementwise_max(a0, zz);
  a1 = __builtin_elementwise_max(a1, zz);

  v8f c0 = {}, c1 = {}, c2 = {}, c3 = {};   // inline-0 C operands
  c0 = wmma_f16(a0, B0, c0);  c0 = wmma_f16(a1, B4, c0);
  c1 = wmma_f16(a0, B1, c1);  c1 = wmma_f16(a1, B5, c1);
  c2 = wmma_f16(a0, B2, c2);  c2 = wmma_f16(a1, B6, c2);
  c3 = wmma_f16(a0, B3, c3);  c3 = wmma_f16(a1, B7, c3);

  if (keep) {     // uniform branch (kmask forced to SGPR by readfirstlane)
    const _Float16* b2r = b2h + s * FEAT;
    // relu clamp deferred to loop exit; packed f16 max-pool
    pooled[0] = __builtin_elementwise_max(pooled[0], cvt_add_h(c0, b2r[0 * 16 + nloc]));
    pooled[1] = __builtin_elementwise_max(pooled[1], cvt_add_h(c1, b2r[1 * 16 + nloc]));
    pooled[2] = __builtin_elementwise_max(pooled[2], cvt_add_h(c2, b2r[2 * 16 + nloc]));
    pooled[3] = __builtin_elementwise_max(pooled[3], cvt_add_h(c3, b2r[3 * 16 + nloc]));
  }
}

// ---------------- weight pre-swizzle kernels ----------------
__global__ __launch_bounds__(256) void swz64_kernel(const float* __restrict__ src,
                                                    _Float16* __restrict__ dst, int nmat) {
  int idx = blockIdx.x * blockDim.x + threadIdx.x;
  if (idx >= nmat * 4096) return;
  int m    = idx >> 12;
  int r    = idx & 4095;
  int f    = r >> 9;
  int lane = (r >> 4) & 31;
  int i    = r & 15;
  int g = (f & 3) * 16 + (lane & 15);
  int k = (f >> 2) * 32 + (lane >> 4) * 8 + (i & 7) + (i >> 3) * 16;
  dst[idx] = (_Float16)src[(size_t)m * 4096 + g * 64 + k];
}

__global__ __launch_bounds__(256) void cvt_rows_kernel(const float* __restrict__ src,
                                                       _Float16* __restrict__ dst, int n) {
  int idx = blockIdx.x * blockDim.x + threadIdx.x;
  if (idx < n) dst[idx] = (_Float16)src[idx];
}

__global__ __launch_bounds__(256) void build_wmls_kernel(const float* __restrict__ Wmu,
                                                         const float* __restrict__ Wls,
                                                         _Float16* __restrict__ dst) {
  int idx = blockIdx.x * blockDim.x + threadIdx.x;
  if (idx >= 1024) return;
  int kw   = idx >> 9;
  int lane = (idx >> 4) & 31;
  int i    = idx & 15;
  int nloc = lane & 15;
  int k = kw * 32 + (lane >> 4) * 8 + (i & 7) + (i >> 3) * 16;
  float v = (nloc == 0) ? Wmu[k] : ((nloc == 1) ? Wls[k] : 0.0f);
  dst[idx] = (_Float16)v;
}

// ---------------- fused main kernel ----------------
// Block: 512 threads = 16 waves = 4 row-tiles x 4 s-quarters. 64 batch rows / block.
__global__ __launch_bounds__(512) void cmp_fused_kernel(
    const float* __restrict__ s_t, const float* __restrict__ a_t,
    const unsigned char* __restrict__ mask_keep,
    const float* __restrict__ ba1, const float* __restrict__ ba2,
    const float* __restrict__ bh1, const float* __restrict__ bh2,
    const float* __restrict__ bmu, const float* __restrict__ bls,
    const _Float16* __restrict__ ws,
    float* __restrict__ out, int Btot)
{
  const int lane  = threadIdx.x & 31;
  const int wave  = threadIdx.x >> 5;
  const int tile  = wave & 3;          // row-tile within block
  const int sq    = wave >> 2;         // s-quarter 0..3
  const int nloc  = lane & 15;
  const int hi    = lane >> 4;
  const int mrow  = hi * 8;
  const int abase = hi * 8;
  const int rowBase = blockIdx.x * 64 + tile * 16;
  const int m = nloc;

  const _Float16* __restrict__ W2h   = ws + OFF_W2;
  const _Float16* __restrict__ Wa1h  = ws + OFF_WA1;
  const _Float16* __restrict__ Wa2h  = ws + OFF_WA2;
  const _Float16* __restrict__ Wh1h  = ws + OFF_WH1;
  const _Float16* __restrict__ Wh2h  = ws + OFF_WH2;
  const _Float16* __restrict__ Wmlsh = ws + OFF_WMLS;
  const _Float16* __restrict__ w1h   = ws + OFF_W1;
  const _Float16* __restrict__ b1h   = ws + OFF_B1;
  const _Float16* __restrict__ b2h   = ws + OFF_B2;

  __shared__ _Float16 xpose[16][16 * LDS_STRIDE];
  _Float16* mylds = &xpose[wave][0];
  _Float16* myrow = &xpose[wave][m * LDS_STRIDE];

  v8h pooled[4];
#pragma unroll
  for (int t = 0; t < 4; ++t)
#pragma unroll
    for (int j = 0; j < 8; ++j) pooled[t][j] = (_Float16)(-60000.0f);

  const int s0 = sq * SCHUNK;

  // 32-bit keep-mask for this wave's s-range (bool bytes are 0/1); force to SGPR
  uint32_t kmask = 0;
  {
    const uint32_t* mw = (const uint32_t*)(mask_keep + s0);
#pragma unroll
    for (int w = 0; w < 8; ++w) {
      uint32_t x = mw[w];
      kmask |= ((x & 1u) | ((x >> 7) & 2u) | ((x >> 14) & 4u) | ((x >> 21) & 8u)) << (4 * w);
    }
  }
  kmask = (uint32_t)__builtin_amdgcn_readfirstlane((int)kmask);

  // ---- per-state-scalar embeds + max-pool over this wave's s-quarter ----
  const float* srow = s_t + (size_t)(rowBase + m) * STATE_DIM;
  const v16h zz = zero16h();
  for (int s = s0; s < s0 + SCHUNK; ++s)
    s_step(s, W2h, w1h, b1h, b2h, srow, (kmask >> (s - s0)) & 1,
           abase, nloc, lane, zz, pooled);

  if (kmask) {   // deferred relu clamp: max_s max(v_s,0) == max(max_s v_s, 0)
    const v8h z8 = zero8h();
#pragma unroll
    for (int t = 0; t < 4; ++t) pooled[t] = __builtin_elementwise_max(pooled[t], z8);
  }

  // ---- action embed MLP (s-quarter 0 waves only; barriers uniform) ----
  if (sq == 0) {
    const float* arow = a_t + (size_t)(rowBase + m) * FEAT;
    v16h a0 = load_afrag_row_f32(arow, 0 + abase);
    v16h a1 = load_afrag_row_f32(arow, 32 + abase);
    layer64_to_lds(a0, a1, Wa1h, ba1, mylds, nloc, hi, mrow, lane);
  }
  __syncthreads();
  if (sq == 0) {
    v16h h0 = load_afrag_h(myrow, 0 + abase);
    v16h h1v = load_afrag_h(myrow, 32 + abase);
    const bool keep_a = mask_keep[STATE_DIM] != 0;
#pragma unroll
    for (int t = 0; t < 4; ++t) {
      v8f c = tile_gemm(h0, h1v, Wa2h, t, lane, ba2[t * 16 + nloc]);
      if (keep_a) {
        v8h ch;
#pragma unroll
        for (int j = 0; j < 8; ++j) ch[j] = (_Float16)fmaxf(c[j], 0.0f);
        pooled[t] = __builtin_elementwise_max(pooled[t], ch);
      }
    }
  }

  // ---- combine the 4 s-quarter partial max-pools (via LDS, f16) ----
  if (sq != 0) {
#pragma unroll
    for (int t = 0; t < 4; ++t)
      *(v8h*)&mylds[((t * 16 + nloc) * 2 + hi) * 8] = pooled[t];
  }
  __syncthreads();
  if (sq == 0) {
#pragma unroll
    for (int p = 1; p < 4; ++p) {
      const _Float16* plds = &xpose[p * 4 + tile][0];
#pragma unroll
      for (int t = 0; t < 4; ++t) {
        v8h pk = *(const v8h*)&plds[((t * 16 + nloc) * 2 + hi) * 8];
        pooled[t] = __builtin_elementwise_max(pooled[t], pk);
      }
    }
  }

  // ---- Gaussian head (s-quarter 0 waves) ----
  if (sq == 0) {
#pragma unroll
    for (int t = 0; t < 4; ++t)
#pragma unroll
      for (int j = 0; j < 8; ++j)
        mylds[(mrow + j) * LDS_STRIDE + t * 16 + nloc] = pooled[t][j];
  }
  __syncthreads();
  if (sq == 0) {
    v16h p0 = load_afrag_h(myrow, 0 + abase);
    v16h p1 = load_afrag_h(myrow, 32 + abase);
    layer64_to_lds(p0, p1, Wh1h, bh1, mylds, nloc, hi, mrow, lane);  // head layer 1
  }
  __syncthreads();
  if (sq == 0) {
    v16h h0 = load_afrag_h(myrow, 0 + abase);
    v16h h1v = load_afrag_h(myrow, 32 + abase);
    layer64_to_lds(h0, h1v, Wh2h, bh2, mylds, nloc, hi, mrow, lane); // head layer 2
  }
  __syncthreads();
  if (sq == 0) {
    v16h h0 = load_afrag_h(myrow, 0 + abase);
    v16h h1v = load_afrag_h(myrow, 32 + abase);
    v16h q0 = *(const v16h*)(Wmlsh + ((size_t)0 * 32 + lane) * 16);
    v16h q1 = *(const v16h*)(Wmlsh + ((size_t)1 * 32 + lane) * 16);
    v8f c;
    const float bv = (nloc == 0) ? bmu[0] : ((nloc == 1) ? bls[0] : 0.0f);
#pragma unroll
    for (int j = 0; j < 8; ++j) c[j] = bv;
    c = wmma_f16(h0, q0, c);
    c = wmma_f16(h1v, q1, c);

    const int row0 = rowBase + mrow;
    if (nloc == 0) {               // mu: column 0 (lanes 0 & 16)
#pragma unroll
      for (int j = 0; j < 8; ++j) out[row0 + j] = c[j];
    } else if (nloc == 1) {        // log_std: column 1 (lanes 1 & 17), clipped
#pragma unroll
      for (int j = 0; j < 8; ++j)
        out[Btot + row0 + j] = fminf(fmaxf(c[j], MIN_LS), MAX_LS);
    }
  }
}

extern "C" void kernel_launch(void* const* d_in, const int* in_sizes, int n_in,
                              void* d_out, int out_size, void* d_ws, size_t ws_size,
                              hipStream_t stream) {
  const float* s_t  = (const float*)d_in[0];
  const float* a_t  = (const float*)d_in[1];
  const unsigned char* mask = (const unsigned char*)d_in[2];
  const float* w1   = (const float*)d_in[3];
  const float* b1   = (const float*)d_in[4];
  const float* W2   = (const float*)d_in[5];
  const float* b2   = (const float*)d_in[6];
  const float* Wa1  = (const float*)d_in[7];
  const float* ba1  = (const float*)d_in[8];
  const float* Wa2  = (const float*)d_in[9];
  const float* ba2  = (const float*)d_in[10];
  const float* Wh1  = (const float*)d_in[11];
  const float* bh1  = (const float*)d_in[12];
  const float* Wh2  = (const float*)d_in[13];
  const float* bh2  = (const float*)d_in[14];
  const float* Wmu  = (const float*)d_in[15];
  const float* bmu  = (const float*)d_in[16];
  const float* Wls  = (const float*)d_in[17];
  const float* bls  = (const float*)d_in[18];
  float* out = (float*)d_out;
  _Float16* ws = (_Float16*)d_ws;   // requires ~1.13 MB of scratch

  const int Btot = in_sizes[0] / STATE_DIM;   // 16384

  // ---- pre-swizzle weights into WMMA fragment layout (f16) ----
  swz64_kernel<<<(128 * 4096) / 256, 256, 0, stream>>>(W2,  ws + OFF_W2, 128);
  swz64_kernel<<<16, 256, 0, stream>>>(Wa1, ws + OFF_WA1, 1);
  swz64_kernel<<<16, 256, 0, stream>>>(Wa2, ws + OFF_WA2, 1);
  swz64_kernel<<<16, 256, 0, stream>>>(Wh1, ws + OFF_WH1, 1);
  swz64_kernel<<<16, 256, 0, stream>>>(Wh2, ws + OFF_WH2, 1);
  build_wmls_kernel<<<4, 256, 0, stream>>>(Wmu, Wls, ws + OFF_WMLS);
  cvt_rows_kernel<<<32, 256, 0, stream>>>(w1, ws + OFF_W1, STATE_DIM * FEAT);
  cvt_rows_kernel<<<32, 256, 0, stream>>>(b1, ws + OFF_B1, STATE_DIM * FEAT);
  cvt_rows_kernel<<<32, 256, 0, stream>>>(b2, ws + OFF_B2, STATE_DIM * FEAT);

  // ---- fused forward ----
  dim3 grid(Btot / 64);    // 256 blocks x 16 waves (4 row-tiles x 4 s-quarters)
  dim3 block(512);
  cmp_fused_kernel<<<grid, block, 0, stream>>>(
      s_t, a_t, mask, ba1, ba2, bh1, bh2, bmu, bls, ws, out, Btot);
}
